// SharedOffsetsPCD_44040594653605
// MI455X (gfx1250) — matrix operand order repaired and checked by reading, not compiled, not run on hardware
//
#include <hip/hip_runtime.h>
#include <math.h>

// ---------------------------------------------------------------------------
// Types for CDNA5 WMMA (wave32, v_wmma_f32_16x16x32_f16, fp32 accumulate)
// ---------------------------------------------------------------------------
typedef _Float16 v16h __attribute__((ext_vector_type(16)));
typedef _Float16 v8h  __attribute__((ext_vector_type(8)));
typedef float    v8f  __attribute__((ext_vector_type(8)));

#define ACT_NONE    0
#define ACT_LRELU   1
#define ACT_SIGMOID 2

__device__ __forceinline__ float apply_act(float v, int act) {
    if (act == ACT_LRELU)   return v >= 0.f ? v : 0.1f * v;
    if (act == ACT_SIGMOID) return 1.f / (1.f + __expf(-v));
    return v;
}

// ---------------------------------------------------------------------------
// Implicit-GEMM 3x3 conv (stride 1, pad 1) with WMMA.
//   GEMM: M = H*W pixels, N = Cout, K = (Cin0+Cin1)*9  (always % 32 == 0)
//   Block = 128 threads (4 waves). Block tile: 64 pixels x 64 Cout.
//   Per K-chunk (32):
//     - 128 threads cooperatively stage a 64x32 f16 A-tile into LDS:
//       each thread owns 1 pixel row x 16 consecutive K (branchless
//       clamp+select zero pad; gathers stay coalesced across lanes,
//       channel decode is wave-uniform), written as 2x ds_store_b128;
//       double-buffered so staging overlaps the WMMAs;
//     - each wave builds one B fragment from 16 contiguous weights
//       (4x global_load_b128, next chunk prefetched at WGP scope) and
//       reuses it across 4 A sub-tiles (2x ds_load_b128 each)
//       -> 4 WMMAs per wave per chunk.
//   fp32 in HBM, f16 into the matrix core, fp32 accumulate, fused bias+act.
// ---------------------------------------------------------------------------
__global__ __launch_bounds__(128)
void conv3x3_wmma(const float* __restrict__ in0, int Cin0,
                  const float* __restrict__ in1, int Cin1,
                  const float* __restrict__ w,   // [Cout, Cin, 3, 3] fp32
                  const float* __restrict__ bias,
                  float* __restrict__ out, int Cout,
                  int H, int W, int act)
{
    // 64 pixel rows x 32 K, rows padded to 40 f16 (80 B) so every 8-f16
    // sub-row is 16B-aligned (b128 LDS ops) with conflict-free striding.
    __shared__ _Float16 aT[2][64][40];

    const int lane = threadIdx.x;            // 0..31
    const int wave = threadIdx.y;            // 0..3
    const int tid  = wave * 32 + lane;       // 0..127
    const int HW   = H * W;
    const int pixBase = blockIdx.x * 64;
    const int b       = blockIdx.z;
    const int Cin     = Cin0 + Cin1;
    const int Ktot    = Cin * 9;

    // Output column (Cout) this lane owns; clamp weight row, store is guarded.
    const int n  = blockIdx.y * 64 + wave * 16 + (lane & 15);
    const int nc = (n < Cout) ? n : (Cout - 1);
    const float* wRow = w + (size_t)nc * Ktot;

    // A layout (matches B): lanes 0-15 K=0..15, lanes 16-31 K=16..31.
    const int aKBase = (lane & 16) ? 16 : 0;
    // B layout: lanes 0-15 rows K=0..15, lanes 16-31 rows K=16..31.
    const int bKBase = (lane & 16) ? 16 : 0;
    const int mLane  = lane & 15;

    // Staging: thread t owns pixel (t&63) and K span [(t>>6)*16, +16).
    // (t>>6) is wave-uniform -> channel decode & pointer select are uniform;
    // lanes map to consecutive pixels -> coalesced global gathers.
    const int stM   = tid & 63;
    const int stKB  = (tid >> 6) * 16;
    const int stPix = pixBase + stM;
    const int stPy  = stPix / W;
    const int stPx  = stPix - stPy * W;

    auto stage = [&](int kk, int pbuf) {
        v8h half0, half1;
#pragma unroll
        for (int e = 0; e < 16; ++e) {
            int kA = kk + stKB + e;
            int c  = kA / 9;
            int t  = kA - c * 9;
            int ty = t / 3;
            int y  = stPy + ty - 1;
            int x  = stPx + (t - ty * 3) - 1;
            // branchless zero padding: clamped unconditional load + select
            int yc = y < 0 ? 0 : (y > H - 1 ? H - 1 : y);
            int xc = x < 0 ? 0 : (x > W - 1 ? W - 1 : x);
            const float* src = (c < Cin0)
                ? in0 + (((size_t)b * Cin0 + c) * HW)
                : in1 + (((size_t)b * Cin1 + (c - Cin0)) * HW);
            float v = src[yc * W + xc];
            v = ((unsigned)y < (unsigned)H && (unsigned)x < (unsigned)W) ? v : 0.f;
            if (e < 8) half0[e] = (_Float16)v;
            else       half1[e - 8] = (_Float16)v;
        }
        // two 16B-aligned LDS stores (ds_store_b128)
        *(v8h*)&aT[pbuf][stM][stKB]     = half0;
        *(v8h*)&aT[pbuf][stM][stKB + 8] = half1;
    };

    v8f acc[4] = { {}, {}, {}, {} };

    stage(0, 0);
    __syncthreads();

    int p = 0;
    for (int kk = 0; kk < Ktot; kk += 32) {
        // ---- B fragment: 16 contiguous fp32 weights -> 4x b128 loads ----
        v16h bm;
        const float4* wp = (const float4*)(wRow + kk + bKBase);
        if (kk + 32 < Ktot)
            __builtin_prefetch(wRow + kk + 32, 0, 3);   // WGP-scope prefetch
#pragma unroll
        for (int j = 0; j < 4; ++j) {
            float4 wv = wp[j];
            bm[4 * j + 0] = (_Float16)wv.x;
            bm[4 * j + 1] = (_Float16)wv.y;
            bm[4 * j + 2] = (_Float16)wv.z;
            bm[4 * j + 3] = (_Float16)wv.w;
        }

        // ---- overlap: stage next chunk into the other LDS buffer ----
        if (kk + 32 < Ktot) stage(kk + 32, p ^ 1);

        // ---- 4 WMMAs reuse bm across the 64-pixel strip ----
#pragma unroll
        for (int s = 0; s < 4; ++s) {
            v8h lo = *(const v8h*)&aT[p][s * 16 + mLane][aKBase];
            v8h hi = *(const v8h*)&aT[p][s * 16 + mLane][aKBase + 8];
            v16h a;
#pragma unroll
            for (int i = 0; i < 8; ++i) { a[i] = lo[i]; a[8 + i] = hi[i]; }
            acc[s] = __builtin_amdgcn_wmma_f32_16x16x32_f16(
                         false, a, false, bm, (short)0, acc[s], false, false);
        }

        __syncthreads();
        p ^= 1;
    }

    if (n < Cout) {
        const float bv   = bias[n];
        const int   mOff = (lane & 16) ? 8 : 0;
#pragma unroll
        for (int s = 0; s < 4; ++s)
#pragma unroll
            for (int r = 0; r < 8; ++r) {
                int m = pixBase + s * 16 + mOff + r;   // D: VGPR r -> M=r (+8 hi)
                out[(((size_t)b * Cout + n) * HW) + m] =
                    apply_act(acc[s][r] + bv, act);
            }
    }
}

// ---------------------------------------------------------------------------
// Modulated deformable 3x3 conv, G=8 groups, Cin=Cout=64, Cg=8.
// Gather-bound: bilinear sampling + small 8x72 grouped matmul per pixel.
// Weights for the block's group staged in LDS.
// ---------------------------------------------------------------------------
__global__ __launch_bounds__(256)
void dconv_kernel(const float* __restrict__ x,
                  const float* __restrict__ off,  int offCtot,
                  const float* __restrict__ mask, int maskCtot,
                  int maskBase, int applySigmoid,
                  const float* __restrict__ w,    // [64, 8, 3, 3] fp32
                  const float* __restrict__ bias,
                  float* __restrict__ out, int H, int W, int act)
{
    __shared__ float wsm[8 * 8 * 9];            // [o][c][k] for this group
    const int g  = blockIdx.y;
    const int b  = blockIdx.z;
    const int HW = H * W;

    for (int t = threadIdx.x; t < 576; t += 256)
        wsm[t] = w[g * 576 + t];
    __syncthreads();

    const int pix = blockIdx.x * 256 + threadIdx.x;
    const int pyi = pix / W;
    const int pxi = pix - pyi * W;

    float acc[8];
#pragma unroll
    for (int o = 0; o < 8; ++o) acc[o] = 0.f;

#pragma unroll
    for (int k = 0; k < 9; ++k) {
        const int gk = g * 9 + k;
        float dy = off[(((size_t)b * offCtot + 2 * gk) * HW) + pix];
        float dx = off[(((size_t)b * offCtot + 2 * gk + 1) * HW) + pix];
        float m  = mask[(((size_t)b * maskCtot + maskBase + gk) * HW) + pix];
        if (applySigmoid) m = 1.f / (1.f + __expf(-m));

        float py = (float)pyi + (float)(k / 3) - 1.f + dy;
        float px = (float)pxi + (float)(k % 3) - 1.f + dx;
        float y0 = floorf(py), x0 = floorf(px);
        float ly = py - y0,   lx = px - x0;
        float wgt[4] = { (1.f - ly) * (1.f - lx), (1.f - ly) * lx,
                         ly * (1.f - lx),         ly * lx };
        int yy[2] = { (int)y0, (int)y0 + 1 };
        int xx[2] = { (int)x0, (int)x0 + 1 };

#pragma unroll
        for (int c = 0; c < 8; ++c) {
            const float* xp = x + (((size_t)b * 64 + g * 8 + c) * HW);
            float s = 0.f;
#pragma unroll
            for (int cy = 0; cy < 2; ++cy)
#pragma unroll
                for (int cx = 0; cx < 2; ++cx) {
                    int Y = yy[cy], X = xx[cx];
                    int Yc = Y < 0 ? 0 : (Y > H - 1 ? H - 1 : Y);
                    int Xc = X < 0 ? 0 : (X > W - 1 ? W - 1 : X);
                    float v = xp[Yc * W + Xc];
                    v = ((unsigned)Y < (unsigned)H && (unsigned)X < (unsigned)W)
                        ? v : 0.f;
                    s += wgt[cy * 2 + cx] * v;
                }
            float col = s * m;
#pragma unroll
            for (int o = 0; o < 8; ++o)
                acc[o] += wsm[(o * 8 + c) * 9 + k] * col;
        }
    }

#pragma unroll
    for (int o = 0; o < 8; ++o) {
        int oc = g * 8 + o;
        out[(((size_t)b * 64 + oc) * HW) + pix] =
            apply_act(acc[o] + bias[oc], act);
    }
}

// ---------------------------------------------------------------------------
// 2x bilinear upsample (half-pixel centers, edge clamp)
// ---------------------------------------------------------------------------
__global__ __launch_bounds__(256)
void up2_kernel(const float* __restrict__ in, float* __restrict__ out,
                int Hin, int Win)
{
    const int Hout = 2 * Hin, Wout = 2 * Win;
    size_t idx = (size_t)blockIdx.x * 256 + threadIdx.x;
    size_t total = (size_t)2 * 64 * Hout * Wout;   // B=2, C=64
    if (idx >= total) return;

    int    xo = (int)(idx % Wout);
    size_t r  = idx / Wout;
    int    yo = (int)(r % Hout);
    size_t bc = r / Hout;

    float sy = (yo + 0.5f) * 0.5f - 0.5f;
    float sx = (xo + 0.5f) * 0.5f - 0.5f;
    float y0f = floorf(sy), x0f = floorf(sx);
    float ly = sy - y0f,    lx = sx - x0f;
    int y0 = (int)y0f, x0 = (int)x0f;
    int y1 = min(Hin - 1, max(0, y0 + 1));
    int x1 = min(Win - 1, max(0, x0 + 1));
    y0 = min(Hin - 1, max(0, y0));
    x0 = min(Win - 1, max(0, x0));

    const float* p = in + bc * (size_t)(Hin * Win);
    float v = (1.f - ly) * ((1.f - lx) * p[y0 * Win + x0] + lx * p[y0 * Win + x1])
            +        ly  * ((1.f - lx) * p[y1 * Win + x0] + lx * p[y1 * Win + x1]);
    out[idx] = v;
}

// ---------------------------------------------------------------------------
// Host-side helpers
// ---------------------------------------------------------------------------
static void conv(const float* in0, int c0, const float* in1, int c1,
                 const float* w, const float* b, float* out, int Cout,
                 int Hh, int act, hipStream_t s)
{
    int HW = Hh * Hh;
    dim3 grid(HW / 64, (Cout + 63) / 64, 2);
    dim3 blk(32, 4);
    conv3x3_wmma<<<grid, blk, 0, s>>>(in0, c0, in1, c1, w, b, out, Cout, Hh, Hh, act);
}

static void dconv(const float* x, const float* off, int offC,
                  const float* mask, int maskC, int maskBase, int sig,
                  const float* w, const float* b, float* out,
                  int Hh, int act, hipStream_t s)
{
    int HW = Hh * Hh;
    dim3 grid(HW / 256, 8, 2);
    dconv_kernel<<<grid, 256, 0, s>>>(x, off, offC, mask, maskC, maskBase, sig,
                                      w, b, out, Hh, Hh, act);
}

static void up2(const float* in, float* out, int Hin, hipStream_t s)
{
    size_t total = (size_t)2 * 64 * (2 * Hin) * (2 * Hin);
    up2_kernel<<<(unsigned)((total + 255) / 256), 256, 0, s>>>(in, out, Hin, Hin);
}

// ---------------------------------------------------------------------------
// Full pipeline (coarse -> fine pyramid + cascade)
// ---------------------------------------------------------------------------
extern "C" void kernel_launch(void* const* d_in, const int* in_sizes, int n_in,
                              void* d_out, int out_size, void* d_ws, size_t ws_size,
                              hipStream_t stream)
{
    (void)in_sizes; (void)n_in; (void)out_size; (void)ws_size;
    const float* in[61];
    for (int i = 0; i < 61; ++i) in[i] = (const float*)d_in[i];

    const float* nrf[3] = { in[0], in[3], in[6] };
    const float* rf [3] = { in[1], in[4], in[7] };
    const float* swf[3] = { in[2], in[5], in[8] };

    const int Hs[3] = { 192, 96, 48 };
    const size_t S[3] = { (size_t)2 * 64 * 192 * 192,
                          (size_t)2 * 64 * 96  * 96,
                          (size_t)2 * 64 * 48  * 48 };

    // workspace layout (fp32): ~191 MB total
    float* wsf = (float*)d_ws;
    size_t o = 0;
    auto alloc = [&](size_t n) { float* p = wsf + o; o += n; return p; };
    float* ofb[3] = { alloc(S[0]), alloc(S[1]), alloc(S[2]) };
    float* dnb[3] = { alloc(S[0]), alloc(S[1]), alloc(S[2]) };
    float* t0 = alloc(S[0]);
    float* t1 = alloc(S[0]);
    float* t2 = alloc(S[0]);
    float* om = alloc((size_t)2 * 216 * 192 * 192);
    float* em = alloc((size_t)2 * 72  * 192 * 192);

    // outputs: (out, shares0, shares1, shares2) concatenated flat
    float* out_f  = (float*)d_out;
    float* share[3];
    share[0] = out_f + S[0];
    share[1] = share[0] + S[0];
    share[2] = share[1] + S[1];

    for (int ii = 0; ii < 3; ++ii) {
        const int i  = 2 - ii;               // coarse -> fine
        const int Hh = Hs[i];
        const int pj = 9 + 12 * i;
        const float *ocf_w = in[pj],      *ocf_b = in[pj + 1];
        const float *ocl_w = in[pj + 2],  *ocl_b = in[pj + 3];
        const float *com_w = in[pj + 4],  *com_b = in[pj + 5];
        const float *em1_w = in[pj + 6],  *em1_b = in[pj + 7];
        const float *em2_w = in[pj + 8],  *em2_b = in[pj + 9];
        const float *dc_w  = in[pj + 10], *dc_b  = in[pj + 11];

        // of = lrelu(conv(cat(nrf, rf)))
        conv(nrf[i], 64, rf[i], 64, ocf_w, ocf_b, t0, 64, Hh, ACT_LRELU, stream);
        const float* ofsrc = t0;
        if (i < 2) {
            up2(ofb[i + 1], t1, Hs[i + 1], stream);
            const float* occ_w = (i == 0) ? in[45] : in[49];
            const float* occ_b = (i == 0) ? in[46] : in[50];
            conv(t0, 64, t1, 64, occ_w, occ_b, t2, 64, Hh, ACT_LRELU, stream);
            ofsrc = t2;
        }
        conv(ofsrc, 64, nullptr, 0, ocl_w, ocl_b, ofb[i], 64, Hh, ACT_LRELU, stream);

        // om = conv(swf) -> [B,216,H,W]: offsets ch 0..143, mask(pre-sig) 144..215
        conv(swf[i], 64, nullptr, 0, com_w, com_b, om, 216, Hh, ACT_NONE, stream);

        // xd = dconv(nrf, offset, sigmoid(mask)); at i==2 fuse lrelu into dn
        if (i == 2)
            dconv(nrf[i], om, 216, om, 216, 144, 1, dc_w, dc_b, dnb[2], Hh, ACT_LRELU, stream);
        else
            dconv(nrf[i], om, 216, om, 216, 144, 1, dc_w, dc_b, t0, Hh, ACT_NONE, stream);

        // em = sigmoid(conv(lrelu(conv(cat(of, swf)))))
        conv(ofb[i], 64, swf[i], 64, em1_w, em1_b, t1, 64, Hh, ACT_LRELU, stream);
        conv(t1, 64, nullptr, 0, em2_w, em2_b, em, 72, Hh, ACT_SIGMOID, stream);

        // shares[i] = dconv(of, offset, em)
        dconv(ofb[i], om, 216, em, 72, 0, 0, dc_w, dc_b, share[i], Hh, ACT_NONE, stream);

        // dn = conv(cat(xd, up2(dn_prev))) for i<2; lrelu at i==1
        if (i < 2) {
            up2(dnb[i + 1], t1, Hs[i + 1], stream);
            const float* fcc_w = (i == 0) ? in[47] : in[51];
            const float* fcc_b = (i == 0) ? in[48] : in[52];
            conv(t0, 64, t1, 64, fcc_w, fcc_b, dnb[i], 64, Hh,
                 (i == 1) ? ACT_LRELU : ACT_NONE, stream);
        }
    }

    // cascade refinement
    conv(dnb[0], 64, rf[0], 64, in[53], in[54], t0, 64, 192, ACT_LRELU, stream);
    conv(t0, 64, nullptr, 0, in[55], in[56], t1, 64, 192, ACT_LRELU, stream);
    conv(t1, 64, nullptr, 0, in[57], in[58], om, 216, 192, ACT_NONE, stream);
    dconv(dnb[0], om, 216, om, 216, 144, 1, in[59], in[60], out_f, 192, ACT_LRELU, stream);
}
